// MessagePassingNN_87110526697909
// MI455X (gfx1250) — matrix operand order repaired and checked by reading, not compiled, not run on hardware
//
#include <hip/hip_runtime.h>
#include <hip/hip_bf16.h>

// ---------------------------------------------------------------------------
// MPNN (2 layers, d=64) for MI455X / gfx1250, wave32, WMMA f16 -> f32.
// ---------------------------------------------------------------------------

typedef __attribute__((ext_vector_type(16))) _Float16 v16h;
typedef __attribute__((ext_vector_type(8)))  _Float16 v8h;
typedef __attribute__((ext_vector_type(8)))  float    v8f;
typedef __attribute__((ext_vector_type(4)))  float    v4f;

union F16x16 { v16h v; v8h h[2]; };

#define DK 64
static constexpr int EF_STRIDE = 136;  // 128 + 8 halves -> conflict-free LDS rows
static constexpr int H_STRIDE  = 72;   // 64 + 8 halves

// Load 8 consecutive f32 and convert to 8 f16 (two 16B global loads).
__device__ inline v8h cvt8(const float* __restrict__ p) {
  v4f x0 = *(const v4f*)p;
  v4f x1 = *(const v4f*)(p + 4);
  v8h r;
#pragma unroll
  for (int i = 0; i < 4; ++i) { r[i] = (_Float16)x0[i]; r[4 + i] = (_Float16)x1[i]; }
  return r;
}

__device__ inline v8f wmma_f16(v16h a, v16h b, v8f c) {
  return __builtin_amdgcn_wmma_f32_16x16x32_f16(false, a, false, b, (short)0, c,
                                                false, false);
}

// ---------------------------------------------------------------------------
// Pack a row-major f32 weight W[K][Nn] into WMMA B-fragments (f16).
// Fragment (kt,nt) per-lane layout: lane<16 holds column nt*16+lane,
// K rows kt*32+{0..7,16..23}; lane>=16 holds same column set, K rows kt*32+{8..15,24..31}.
// Stored contiguously: out[((kt*NT+nt)*32 + lane)*16 + j].
// ---------------------------------------------------------------------------
__global__ void pack_wmma_b_kernel(const float* __restrict__ W,
                                   _Float16* __restrict__ out, int K, int Nn) {
  const int t = blockIdx.x * blockDim.x + threadIdx.x;
  const int NT = Nn >> 4;
  const int nfrag = (K >> 5) * NT;
  if (t >= nfrag * 32) return;
  const int frag = t >> 5, lane = t & 31;
  const int nt = frag % NT;
  const int kt = frag / NT;
  const int n = nt * 16 + (lane & 15);
  const int kbase = kt * 32 + (lane >> 4) * 8;
  _Float16* o = out + (size_t)t * 16;
#pragma unroll
  for (int j = 0; j < 16; ++j)
    o[j] = (_Float16)W[(size_t)(kbase + (j < 8 ? j : j + 8)) * Nn + n];
}

// ---------------------------------------------------------------------------
// Edge MLP + scatter-add aggregation.
//   msg = tanh( relu( [x[src]|x[dst]] @ W1 + b1 ) @ W2 + b2 )  -> atomicAdd agg[dst]
// Block: 256 thr = 2 groups x 4 waves. Group handles one 16-edge tile;
// wave owns one 16-col output tile (nt) and keeps its weight frags in VGPRs.
// ---------------------------------------------------------------------------
__global__ __launch_bounds__(256)
void edge_mlp_kernel(const float* __restrict__ x, const int* __restrict__ src,
                     const int* __restrict__ dst,
                     const _Float16* __restrict__ w1p, const float* __restrict__ b1,
                     const _Float16* __restrict__ w2p, const float* __restrict__ b2,
                     float* __restrict__ agg, int E) {
  __shared__ __align__(16) _Float16 ef[2][16][EF_STRIDE];  // [group][edge][128 cols]
  __shared__ __align__(16) _Float16 hb[2][16][H_STRIDE];   // hidden after relu
  __shared__ int ddst[2][16];

  const int tid  = threadIdx.x;
  const int wave = tid >> 5;
  const int lane = tid & 31;
  const int g    = wave >> 2;   // group 0/1
  const int nt   = wave & 3;    // output column tile
  const int hi   = lane >> 4;   // lane half
  const int lm   = lane & 15;

  // Per-wave weight fragments live in registers for the whole kernel.
  v16h w1f[4], w2f[2];
#pragma unroll
  for (int kt = 0; kt < 4; ++kt)
    w1f[kt] = *(const v16h*)(w1p + ((size_t)((kt * 4 + nt) * 32 + lane)) * 16);
#pragma unroll
  for (int kt = 0; kt < 2; ++kt)
    w2f[kt] = *(const v16h*)(w2p + ((size_t)((kt * 4 + nt) * 32 + lane)) * 16);

  const float bias1 = b1[nt * 16 + lm];
  const float bias2 = b2[nt * 16 + lm];

  const int ntile = (E + 15) >> 4;
  const int npair = (ntile + 1) >> 1;

  for (int p = blockIdx.x; p < npair; p += gridDim.x) {
    const int tile  = p * 2 + g;
    const bool tv   = tile < ntile;         // uniform within the group
    const int ebase = tile << 4;

    // ---- stage [x[src]|x[dst]] tile -> f16 LDS (128 threads per group) ----
    {
      const int tg = tid & 127;
      const int r = tg >> 3;   // edge row 0..15
      const int c = tg & 7;    // 16-float chunk 0..7
      if (tv) {
        const int e = ebase + r;
        _Float16* o = &ef[g][r][c * 16];
        if (e < E) {
          const int sn = src[e];
          const int dn = dst[e];
          if (c == 0) ddst[g][r] = dn;
          const float* rp = (c < 4) ? (x + (size_t)sn * DK + c * 16)
                                    : (x + (size_t)dn * DK + (c - 4) * 16);
          *(v8h*)(o)     = cvt8(rp);
          *(v8h*)(o + 8) = cvt8(rp + 8);
        } else {
          v8h z = {};
          *(v8h*)(o) = z; *(v8h*)(o + 8) = z;
          if (c == 0) ddst[g][r] = 0;
        }
      }
    }
    __syncthreads();

    // ---- GEMM1: [16x128] @ [128x64] (this wave: 16 cols), relu -> LDS ----
    if (tv) {
      v8f acc = {};
#pragma unroll
      for (int kt = 0; kt < 4; ++kt) {
        F16x16 a;
        const _Float16* ap = &ef[g][lm][kt * 32 + hi * 8];
        a.h[0] = *(const v8h*)ap;
        a.h[1] = *(const v8h*)(ap + 16);
        acc = wmma_f16(a.v, w1f[kt], acc);
      }
#pragma unroll
      for (int r = 0; r < 8; ++r) {
        float v = acc[r] + bias1;
        v = v > 0.f ? v : 0.f;
        hb[g][r + hi * 8][nt * 16 + lm] = (_Float16)v;
      }
    }
    __syncthreads();

    // ---- GEMM2: [16x64] @ [64x64], tanh, atomic scatter to agg[dst] ----
    if (tv) {
      v8f acc = {};
#pragma unroll
      for (int kt = 0; kt < 2; ++kt) {
        F16x16 a;
        const _Float16* ap = &hb[g][lm][kt * 32 + hi * 8];
        a.h[0] = *(const v8h*)ap;
        a.h[1] = *(const v8h*)(ap + 16);
        acc = wmma_f16(a.v, w2f[kt], acc);
      }
#pragma unroll
      for (int r = 0; r < 8; ++r) {
        const int m = r + hi * 8;
        const int e = ebase + m;
        if (e < E) {
          const float v = tanhf(acc[r] + bias2);
          atomicAdd(&agg[(size_t)ddst[g][m] * DK + nt * 16 + lm], v);
        }
      }
    }
    __syncthreads();  // protect ddst/ef for next iteration's stage
  }
}

// ---------------------------------------------------------------------------
// Node MLPs: out[:, 0:32] = mlp2(agg, n1W*), out[:, 32:64] = mlp2(xin, n2W*).
// One wave per 16-node tile; intra-wave LDS ordering via s_wait_dscnt.
// ---------------------------------------------------------------------------
__global__ __launch_bounds__(256)
void node_mlp_kernel(const float* __restrict__ agg, const float* __restrict__ xin,
                     const _Float16* __restrict__ n1w1p, const float* __restrict__ n1b1,
                     const _Float16* __restrict__ n1w2p, const float* __restrict__ n1b2,
                     const _Float16* __restrict__ n2w1p, const float* __restrict__ n2b1,
                     const _Float16* __restrict__ n2w2p, const float* __restrict__ n2b2,
                     float* __restrict__ out, int N) {
  __shared__ __align__(16) _Float16 ht[8][16][H_STRIDE];
  const int tid = threadIdx.x, wave = tid >> 5, lane = tid & 31;
  const int hi = lane >> 4, lm = lane & 15;
  const int ntile = (N + 15) >> 4;

  for (int tile = blockIdx.x * 8 + wave; tile < ntile; tile += gridDim.x * 8) {
    const int nb = tile << 4;
#pragma unroll
    for (int path = 0; path < 2; ++path) {
      const float*    in = path ? xin   : agg;
      const _Float16* W1 = path ? n2w1p : n1w1p;
      const float*    B1 = path ? n2b1  : n1b1;
      const _Float16* W2 = path ? n2w2p : n1w2p;
      const float*    B2 = path ? n2b2  : n1b2;

      // GEMM1: [16x64] @ [64x64] -> relu -> LDS (f16)
      for (int nt = 0; nt < 4; ++nt) {
        v8f acc = {};
#pragma unroll
        for (int kt = 0; kt < 2; ++kt) {
          int node = nb + lm;
          if (node >= N) node = N - 1;  // clamp: garbage rows never stored
          const float* rp = in + (size_t)node * DK + kt * 32 + hi * 8;
          F16x16 a;
          a.h[0] = cvt8(rp);
          a.h[1] = cvt8(rp + 16);
          v16h b = *(const v16h*)(W1 + ((size_t)((kt * 4 + nt) * 32 + lane)) * 16);
          acc = wmma_f16(a.v, b, acc);
        }
        const float bias = B1[nt * 16 + lm];
#pragma unroll
        for (int r = 0; r < 8; ++r) {
          float v = acc[r] + bias;
          v = v > 0.f ? v : 0.f;
          ht[wave][r + hi * 8][nt * 16 + lm] = (_Float16)v;
        }
      }
      asm volatile("s_wait_dscnt 0" ::: "memory");  // LDS RAW within wave

      // GEMM2: [16x64] @ [64x32] -> tanh -> out columns [path*32, path*32+32)
      for (int nt = 0; nt < 2; ++nt) {
        v8f acc = {};
#pragma unroll
        for (int kt = 0; kt < 2; ++kt) {
          F16x16 a;
          const _Float16* ap = &ht[wave][lm][kt * 32 + hi * 8];
          a.h[0] = *(const v8h*)ap;
          a.h[1] = *(const v8h*)(ap + 16);
          v16h b = *(const v16h*)(W2 + ((size_t)((kt * 2 + nt) * 32 + lane)) * 16);
          acc = wmma_f16(a.v, b, acc);
        }
        const float bias = B2[nt * 16 + lm];
#pragma unroll
        for (int r = 0; r < 8; ++r) {
          const int node = nb + r + hi * 8;
          if (node < N)
            out[(size_t)node * DK + path * 32 + nt * 16 + lm] = tanhf(acc[r] + bias);
        }
      }
      asm volatile("s_wait_dscnt 0" ::: "memory");  // before next path reuses ht
    }
  }
}

// ---------------------------------------------------------------------------
// In-place ReLU + LayerNorm over rows of 64. One wave per row (lane -> 2 cols).
// ---------------------------------------------------------------------------
__global__ __launch_bounds__(256)
void relu_ln_kernel(float* __restrict__ h, const float* __restrict__ g,
                    const float* __restrict__ b, int N) {
  const int row = blockIdx.x * 8 + (threadIdx.x >> 5);
  const int lane = threadIdx.x & 31;
  if (row >= N) return;
  float* rp = h + (size_t)row * DK;
  float v0 = rp[lane];      v0 = v0 > 0.f ? v0 : 0.f;
  float v1 = rp[lane + 32]; v1 = v1 > 0.f ? v1 : 0.f;
  float s = v0 + v1, ss = v0 * v0 + v1 * v1;
#pragma unroll
  for (int m = 16; m >= 1; m >>= 1) {
    s  += __shfl_xor(s, m, 32);
    ss += __shfl_xor(ss, m, 32);
  }
  const float mu  = s * (1.f / 64.f);
  const float var = ss * (1.f / 64.f) - mu * mu;
  const float inv = rsqrtf(var + 1e-5f);
  rp[lane]      = (v0 - mu) * inv * g[lane]      + b[lane];
  rp[lane + 32] = (v1 - mu) * inv * g[lane + 32] + b[lane + 32];
}

// ---------------------------------------------------------------------------
// Host-side orchestration.
// ---------------------------------------------------------------------------
// Packed-weight offsets (in halves) within one layer's pack region:
static constexpr size_t PK_EW1  = 0;                   // 128x64 -> 8192
static constexpr size_t PK_EW2  = PK_EW1 + 128 * 64;   //  64x64 -> 4096
static constexpr size_t PK_N1W1 = PK_EW2 + 64 * 64;
static constexpr size_t PK_N1W2 = PK_N1W1 + 64 * 64;   //  64x32 -> 2048
static constexpr size_t PK_N2W1 = PK_N1W2 + 64 * 32;
static constexpr size_t PK_N2W2 = PK_N2W1 + 64 * 64;
static constexpr size_t PK_LAYER = PK_N2W2 + 64 * 32;  // 24576 halves / layer

extern "C" void kernel_launch(void* const* d_in, const int* in_sizes, int n_in,
                              void* d_out, int out_size, void* d_ws, size_t ws_size,
                              hipStream_t stream) {
  const float* x   = (const float*)d_in[0];
  const int*   src = (const int*)d_in[1];
  const int*   dst = (const int*)d_in[2];
  // Layer param dicts flatten in insertion order:
  //   eW1 eb1 eW2 eb2 n1W1 n1b1 n1W2 n1b2 n2W1 n2b1 n2W2 n2b2
  const float* L[2][12];
  for (int i = 0; i < 12; ++i) {
    L[0][i] = (const float*)d_in[3 + i];
    L[1][i] = (const float*)d_in[15 + i];
  }
  const float* ln_g = (const float*)d_in[27];
  const float* ln_b = (const float*)d_in[28];

  const int N = in_sizes[0] / DK;
  const int E = in_sizes[1];

  float* agg  = (float*)d_ws;
  float* hbuf = agg + (size_t)N * DK;
  _Float16* pk = (_Float16*)(hbuf + (size_t)N * DK);

  auto pack = [&](const float* W, _Float16* o, int K, int Nn) {
    const int threads = (K / 32) * (Nn / 16) * 32;
    pack_wmma_b_kernel<<<(threads + 255) / 256, 256, 0, stream>>>(W, o, K, Nn);
  };

  for (int l = 0; l < 2; ++l) {
    _Float16* base = pk + (size_t)l * PK_LAYER;
    pack(L[l][0],  base + PK_EW1,  128, 64);  // eW1
    pack(L[l][2],  base + PK_EW2,   64, 64);  // eW2
    pack(L[l][4],  base + PK_N1W1,  64, 64);  // n1W1
    pack(L[l][6],  base + PK_N1W2,  64, 32);  // n1W2
    pack(L[l][8],  base + PK_N2W1,  64, 64);  // n2W1
    pack(L[l][10], base + PK_N2W2,  64, 32);  // n2W2
  }

  const int ntile_e = (E + 15) / 16;
  const int npair   = (ntile_e + 1) / 2;
  const int grid_e  = npair < 8192 ? npair : 8192;
  const int grid_n  = ((N + 15) / 16 + 7) / 8;
  const int grid_ln = (N + 7) / 8;

  for (int l = 0; l < 2; ++l) {
    const float* feat = (l == 0) ? x : hbuf;
    float* nout = (l == 0) ? hbuf : (float*)d_out;
    _Float16* base = pk + (size_t)l * PK_LAYER;

    hipMemsetAsync(agg, 0, (size_t)N * DK * sizeof(float), stream);
    edge_mlp_kernel<<<grid_e, 256, 0, stream>>>(
        feat, src, dst, base + PK_EW1, L[l][1], base + PK_EW2, L[l][3], agg, E);
    node_mlp_kernel<<<grid_n, 256, 0, stream>>>(
        agg, feat, base + PK_N1W1, L[l][5], base + PK_N1W2, L[l][7],
        base + PK_N2W1, L[l][9], base + PK_N2W2, L[l][11], nout, N);
    if (l == 0)
      relu_ln_kernel<<<grid_ln, 256, 0, stream>>>(hbuf, ln_g, ln_b, N);
  }
}
